// R_HGNN_Layer_58342835749543
// MI455X (gfx1250) — matrix operand order repaired and checked by reading, not compiled, not run on hardware
//
#include <hip/hip_runtime.h>

// ---- problem constants (match reference) ----
#define N_NODES 100000
#define E_EDGES 500000
#define H_HEADS 8
#define D_DIM   32
#define IN_DIM  32
#define HD      256          // H*D
#define SLOPE   0.2f

typedef __attribute__((ext_vector_type(16))) _Float16 v16h;
typedef __attribute__((ext_vector_type(8)))  float    v8f;

__device__ __forceinline__ float leaky_f(float x) { return x >= 0.f ? x : SLOPE * x; }
__device__ __forceinline__ float neg_inf_f() { return -__builtin_inff(); }

// ---------------------------------------------------------------------------
// Tiny kernel: ra[r][512] = rel_emb[r] @ W_rel[r]   (attention vectors)
//              rel_out[r][256] = rel_emb[r] @ W_prop[r] + b_prop[r]
// ---------------------------------------------------------------------------
__global__ void ra_kernel(const float* __restrict__ rel_emb,
                          const float* __restrict__ W_rel,
                          const float* __restrict__ W_prop,
                          const float* __restrict__ b_prop,
                          float* __restrict__ ra,
                          float* __restrict__ rel_out) {
    const int r = blockIdx.x;
    const int j = threadIdx.x;                 // 0..511
    float s = 0.f;
    #pragma unroll
    for (int i = 0; i < IN_DIM; ++i)
        s += rel_emb[r * IN_DIM + i] * W_rel[((size_t)r * IN_DIM + i) * (2 * HD) + j];
    ra[r * (2 * HD) + j] = s;
    if (j < HD) {
        float p = 0.f;
        #pragma unroll
        for (int i = 0; i < IN_DIM; ++i)
            p += rel_emb[r * IN_DIM + i] * W_prop[((size_t)r * IN_DIM + i) * HD + j];
        rel_out[r * HD + j] = p + b_prop[r * HD + j];
    }
}

// ---------------------------------------------------------------------------
// WMMA projection: proj[r] = feat[r] @ W_node[dt(r)]   ([N,32] @ [32,256])
// One v_wmma_f32_16x16x32_f16 per 16x16 output tile (K=32 exactly).
// Also zeroes the message accumulator tile (per-call re-init for determinism).
// ---------------------------------------------------------------------------
__global__ void proj_kernel(const float* __restrict__ feat,
                            const float* __restrict__ W_node,
                            float* __restrict__ proj,
                            float* __restrict__ acc) {
    const int r    = blockIdx.z;
    const int dt   = 1 - (r & 1);              // RELS dst type
    const int wave = threadIdx.x >> 5;
    const int lane = threadIdx.x & 31;
    const int tile = blockIdx.x * 8 + wave;    // 100000 tiles / relation
    const int tileM = tile >> 4;               // 6250 row tiles
    const int tileN = tile & 15;               // 16 col tiles
    const int idx16 = lane & 15;
    const int half  = lane >> 4;

    // A operand: 16x32 f16, lane holds row (lane&15); elems 0..7 -> K=8*half..,
    // elems 8..15 -> K=16+8*half..   (ISA 7.12.2 16-bit A layout)
    const float* Arow = feat + ((size_t)r * N_NODES + (size_t)tileM * 16 + idx16) * IN_DIM;
    v16h a;
    {
        const float4* A4 = (const float4*)(Arow + 8 * half);
        float4 lo0 = A4[0], lo1 = A4[1];
        const float4* B4 = (const float4*)(Arow + 16 + 8 * half);
        float4 hi0 = B4[0], hi1 = B4[1];
        a[0]=(_Float16)lo0.x; a[1]=(_Float16)lo0.y; a[2]=(_Float16)lo0.z; a[3]=(_Float16)lo0.w;
        a[4]=(_Float16)lo1.x; a[5]=(_Float16)lo1.y; a[6]=(_Float16)lo1.z; a[7]=(_Float16)lo1.w;
        a[8]=(_Float16)hi0.x; a[9]=(_Float16)hi0.y; a[10]=(_Float16)hi0.z; a[11]=(_Float16)hi0.w;
        a[12]=(_Float16)hi1.x; a[13]=(_Float16)hi1.y; a[14]=(_Float16)hi1.z; a[15]=(_Float16)hi1.w;
    }

    // B operand: 32x16 f16, lane holds col (lane&15); elem e -> K = 8*half + e (+8 if e>=8)
    const int gcol = tileN * 16 + idx16;
    const float* B = W_node + (size_t)dt * IN_DIM * HD;
    v16h b;
    #pragma unroll
    for (int e = 0; e < 16; ++e)
        b[e] = (_Float16)B[(8 * half + e + (e >= 8 ? 8 : 0)) * HD + gcol];

    v8f c = {};
    c = __builtin_amdgcn_wmma_f32_16x16x32_f16(false, a, false, b, (short)0, c, false, false);

    float* P = proj + (size_t)r * N_NODES * HD;
    float* Z = acc  + (size_t)r * N_NODES * HD;
    #pragma unroll
    for (int v = 0; v < 8; ++v) {
        const int gm = tileM * 16 + 8 * half + v;   // C/D layout: m = v + 8*(lane>=16)
        P[(size_t)gm * HD + gcol] = c[v];
        Z[(size_t)gm * HD + gcol] = 0.f;            // re-init accumulator every call
    }
}

// ---------------------------------------------------------------------------
// Per-node attention scores: e_src (uses proj[REV[r]], ra high half),
// e_dst (uses proj[r], ra low half); also init segment max / sum.
// ---------------------------------------------------------------------------
__global__ void scores_kernel(const float* __restrict__ proj,
                              const float* __restrict__ ra,
                              float* __restrict__ e_src, float* __restrict__ e_dst,
                              float* __restrict__ mmax, float* __restrict__ zsum) {
    const int r   = blockIdx.z;
    const int rev = r ^ 1;                                 // REV = [1,0,3,2]
    const int idx = blockIdx.x * blockDim.x + threadIdx.x; // n*8 + h
    if (idx >= N_NODES * H_HEADS) return;
    const int n = idx >> 3, h = idx & 7;
    const float* fs = proj + ((size_t)rev * N_NODES + n) * HD + h * D_DIM;
    const float* fd = proj + ((size_t)r   * N_NODES + n) * HD + h * D_DIM;
    const float* rah = ra + r * (2 * HD) + h * (2 * D_DIM);
    float es = 0.f, ed = 0.f;
    #pragma unroll
    for (int d = 0; d < D_DIM; ++d) {
        ed += fd[d] * rah[d];
        es += fs[d] * rah[D_DIM + d];
    }
    const size_t o = (size_t)r * N_NODES * H_HEADS + idx;
    e_src[o] = es;
    e_dst[o] = ed;
    mmax[o]  = neg_inf_f();
    zsum[o]  = 0.f;
}

// ---------------------------------------------------------------------------
// Edge pass 1: segment max per (dst node, head)
// ---------------------------------------------------------------------------
__global__ void edge_max_kernel(const int* __restrict__ src, const int* __restrict__ dst,
                                const float* __restrict__ e_src, const float* __restrict__ e_dst,
                                float* __restrict__ mmax) {
    const int r   = blockIdx.z;
    const int idx = blockIdx.x * blockDim.x + threadIdx.x; // e*8 + h
    if (idx >= E_EDGES * H_HEADS) return;
    const int e = idx >> 3, h = idx & 7;
    const int s = src[(size_t)r * E_EDGES + e];
    const int d = dst[(size_t)r * E_EDGES + e];
    const size_t rb = (size_t)r * N_NODES * H_HEADS;
    const float v = leaky_f(e_src[rb + s * 8 + h] + e_dst[rb + d * 8 + h]);
    atomicMax(&mmax[rb + d * 8 + h], v);
}

// m = isfinite(m) ? m : 0   (nodes with no in-edges)
__global__ void fix_max_kernel(float* __restrict__ mmax) {
    const size_t idx = (size_t)blockIdx.x * blockDim.x + threadIdx.x;
    if (idx >= (size_t)4 * N_NODES * H_HEADS) return;
    const float m = mmax[idx];
    if (m == neg_inf_f()) mmax[idx] = 0.f;
}

// ---------------------------------------------------------------------------
// Edge pass 2: z = segment_sum(exp(e - m[d]))
// ---------------------------------------------------------------------------
__global__ void edge_sum_kernel(const int* __restrict__ src, const int* __restrict__ dst,
                                const float* __restrict__ e_src, const float* __restrict__ e_dst,
                                const float* __restrict__ mmax, float* __restrict__ zsum) {
    const int r   = blockIdx.z;
    const int idx = blockIdx.x * blockDim.x + threadIdx.x;
    if (idx >= E_EDGES * H_HEADS) return;
    const int e = idx >> 3, h = idx & 7;
    const int s = src[(size_t)r * E_EDGES + e];
    const int d = dst[(size_t)r * E_EDGES + e];
    const size_t rb = (size_t)r * N_NODES * H_HEADS;
    const float v  = leaky_f(e_src[rb + s * 8 + h] + e_dst[rb + d * 8 + h]);
    const float ex = __expf(v - mmax[rb + d * 8 + h]);
    atomicAdd(&zsum[rb + d * 8 + h], ex);
}

// ---------------------------------------------------------------------------
// Edge pass 3: acc[d] += fs[s] * a  — one wave per edge; lane owns 8
// contiguous floats of the 256-wide message (head = lane/4).
// ---------------------------------------------------------------------------
__global__ void edge_msg_kernel(const int* __restrict__ src, const int* __restrict__ dst,
                                const float* __restrict__ proj,
                                const float* __restrict__ e_src, const float* __restrict__ e_dst,
                                const float* __restrict__ mmax, const float* __restrict__ zsum,
                                float* __restrict__ acc) {
    const int r    = blockIdx.z;
    const int rev  = r ^ 1;
    const int wave = threadIdx.x >> 5;
    const int lane = threadIdx.x & 31;
    const int e    = blockIdx.x * 8 + wave;
    if (e >= E_EDGES) return;
    const int s = src[(size_t)r * E_EDGES + e];
    const int d = dst[(size_t)r * E_EDGES + e];
    const int h = lane >> 2;
    const size_t rb = (size_t)r * N_NODES * H_HEADS;
    const float v = leaky_f(e_src[rb + s * 8 + h] + e_dst[rb + d * 8 + h]);
    const float a = __expf(v - mmax[rb + d * 8 + h]) / zsum[rb + d * 8 + h];
    const float* fs = proj + ((size_t)rev * N_NODES + s) * HD + lane * 8;
    float*       A  = acc  + ((size_t)r   * N_NODES + d) * HD + lane * 8;
    #pragma unroll
    for (int j = 0; j < 8; ++j)
        atomicAdd(&A[j], fs[j] * a);
}

// ---------------------------------------------------------------------------
// Finalize (WMMA): res = feat[r] @ W_res[dt] + b_res[dt];
// outs[r] = relu(acc)*sig(alpha[dt]) + res*(1-sig) — in place over acc.
// ---------------------------------------------------------------------------
__global__ void finalize_kernel(const float* __restrict__ feat,
                                const float* __restrict__ W_res,
                                const float* __restrict__ b_res,
                                const float* __restrict__ alpha,
                                float* __restrict__ acc) {
    const int r    = blockIdx.z;
    const int dt   = 1 - (r & 1);
    const int wave = threadIdx.x >> 5;
    const int lane = threadIdx.x & 31;
    const int tile = blockIdx.x * 8 + wave;
    const int tileM = tile >> 4;
    const int tileN = tile & 15;
    const int idx16 = lane & 15;
    const int half  = lane >> 4;

    const float sig = 1.f / (1.f + __expf(-alpha[dt]));

    const float* Arow = feat + ((size_t)r * N_NODES + (size_t)tileM * 16 + idx16) * IN_DIM;
    v16h a;
    {
        const float4* A4 = (const float4*)(Arow + 8 * half);
        float4 lo0 = A4[0], lo1 = A4[1];
        const float4* B4 = (const float4*)(Arow + 16 + 8 * half);
        float4 hi0 = B4[0], hi1 = B4[1];
        a[0]=(_Float16)lo0.x; a[1]=(_Float16)lo0.y; a[2]=(_Float16)lo0.z; a[3]=(_Float16)lo0.w;
        a[4]=(_Float16)lo1.x; a[5]=(_Float16)lo1.y; a[6]=(_Float16)lo1.z; a[7]=(_Float16)lo1.w;
        a[8]=(_Float16)hi0.x; a[9]=(_Float16)hi0.y; a[10]=(_Float16)hi0.z; a[11]=(_Float16)hi0.w;
        a[12]=(_Float16)hi1.x; a[13]=(_Float16)hi1.y; a[14]=(_Float16)hi1.z; a[15]=(_Float16)hi1.w;
    }

    const int gcol = tileN * 16 + idx16;
    const float* B = W_res + (size_t)dt * IN_DIM * HD;
    v16h b;
    #pragma unroll
    for (int e = 0; e < 16; ++e)
        b[e] = (_Float16)B[(8 * half + e + (e >= 8 ? 8 : 0)) * HD + gcol];

    v8f c = {};
    c = __builtin_amdgcn_wmma_f32_16x16x32_f16(false, a, false, b, (short)0, c, false, false);

    const float bias = b_res[dt * HD + gcol];
    float* O = acc + (size_t)r * N_NODES * HD;
    #pragma unroll
    for (int v = 0; v < 8; ++v) {
        const int gm = tileM * 16 + 8 * half + v;
        const size_t o = (size_t)gm * HD + gcol;
        float msg = O[o];
        msg = msg > 0.f ? msg : 0.f;                 // relu
        const float res = c[v] + bias;
        O[o] = msg * sig + res * (1.f - sig);
    }
}

// ---------------------------------------------------------------------------
// Relation crossing: 2-way leaky-softmax over relations per (node, head).
// t=0 -> cross_A from outs[1],outs[3]; t=1 -> cross_B from outs[0],outs[2].
// ---------------------------------------------------------------------------
__global__ void crossing_kernel(const float* __restrict__ outs,
                                const float* __restrict__ Wx,
                                float* __restrict__ out) {
    const int t   = blockIdx.z;
    const int idx = blockIdx.x * blockDim.x + threadIdx.x; // n*8 + h
    if (idx >= N_NODES * H_HEADS) return;
    const int n = idx >> 3, h = idx & 7;
    const int r0 = (t == 0) ? 1 : 0;
    const int r1 = (t == 0) ? 3 : 2;
    const float* f0 = outs + ((size_t)r0 * N_NODES + n) * HD + h * D_DIM;
    const float* f1 = outs + ((size_t)r1 * N_NODES + n) * HD + h * D_DIM;
    const float* w  = Wx + t * HD + h * D_DIM;
    float a0 = 0.f, a1 = 0.f;
    #pragma unroll
    for (int d = 0; d < D_DIM; ++d) { a0 += f0[d] * w[d]; a1 += f1[d] * w[d]; }
    a0 = leaky_f(a0); a1 = leaky_f(a1);
    const float mx = fmaxf(a0, a1);
    float p0 = __expf(a0 - mx), p1 = __expf(a1 - mx);
    const float inv = 1.f / (p0 + p1);
    p0 *= inv; p1 *= inv;
    float* o = out + ((size_t)t * N_NODES + n) * HD + h * D_DIM;
    #pragma unroll
    for (int d = 0; d < D_DIM; ++d)
        o[d] = f0[d] * p0 + f1[d] * p1;
}

// ---------------------------------------------------------------------------
extern "C" void kernel_launch(void* const* d_in, const int* in_sizes, int n_in,
                              void* d_out, int out_size, void* d_ws, size_t ws_size,
                              hipStream_t stream) {
    const float* feat    = (const float*)d_in[0];
    const float* rel_emb = (const float*)d_in[1];
    const float* W_node  = (const float*)d_in[2];
    const float* W_rel   = (const float*)d_in[3];
    const float* W_prop  = (const float*)d_in[4];
    const float* b_prop  = (const float*)d_in[5];
    const float* W_res   = (const float*)d_in[6];
    const float* b_res   = (const float*)d_in[7];
    const float* alpha   = (const float*)d_in[8];
    const float* Wx      = (const float*)d_in[9];
    const int*   src     = (const int*)d_in[10];
    const int*   dst     = (const int*)d_in[11];
    float* out = (float*)d_out;

    // workspace layout (floats)
    float* ws = (float*)d_ws;
    const size_t PROJ = (size_t)4 * N_NODES * HD;   // 102.4M floats each
    const size_t NH   = (size_t)4 * N_NODES * H_HEADS;
    float* proj  = ws;
    float* acc   = proj  + PROJ;      // accumulator, becomes outs in-place
    float* e_src = acc   + PROJ;
    float* e_dst = e_src + NH;
    float* mmax  = e_dst + NH;
    float* zsum  = mmax  + NH;
    float* ra    = zsum  + NH;        // 4*512 floats

    // 0) relation attention vectors + rel_out (direct to output tail)
    ra_kernel<<<dim3(4), dim3(512), 0, stream>>>(
        rel_emb, W_rel, W_prop, b_prop, ra, out + (size_t)2 * N_NODES * HD);

    // 1) WMMA projections (+ zero accumulator)
    proj_kernel<<<dim3(12500, 1, 4), dim3(256), 0, stream>>>(feat, W_node, proj, acc);

    // 2) per-node scores (+ init max/sum)
    scores_kernel<<<dim3((N_NODES * H_HEADS + 255) / 256, 1, 4), dim3(256), 0, stream>>>(
        proj, ra, e_src, e_dst, mmax, zsum);

    // 3) edge softmax: max, fix -inf, sum
    edge_max_kernel<<<dim3((E_EDGES * H_HEADS + 255) / 256, 1, 4), dim3(256), 0, stream>>>(
        src, dst, e_src, e_dst, mmax);
    fix_max_kernel<<<dim3((4 * N_NODES * H_HEADS + 255) / 256), dim3(256), 0, stream>>>(mmax);
    edge_sum_kernel<<<dim3((E_EDGES * H_HEADS + 255) / 256, 1, 4), dim3(256), 0, stream>>>(
        src, dst, e_src, e_dst, mmax, zsum);

    // 4) message scatter (wave per edge)
    edge_msg_kernel<<<dim3(E_EDGES / 8, 1, 4), dim3(256), 0, stream>>>(
        src, dst, proj, e_src, e_dst, mmax, zsum, acc);

    // 5) WMMA residual + sigmoid mix (in place over acc -> outs)
    finalize_kernel<<<dim3(12500, 1, 4), dim3(256), 0, stream>>>(
        feat, W_res, b_res, alpha, acc);

    // 6) relation crossing -> node_out
    crossing_kernel<<<dim3((N_NODES * H_HEADS + 255) / 256, 1, 2), dim3(256), 0, stream>>>(
        acc, Wx, out);
}
